// SentenceEnforcedSwitchMoeBlock_44667659878789
// MI455X (gfx1250) — matrix operand order repaired and checked by reading, not compiled, not run on hardware
//
#include <hip/hip_runtime.h>
#include <math.h>

// ---------------------------------------------------------------------------
// Switch-MoE FFN for MI455X (gfx1250, wave32).
//   stage1: H1 = gelu(X[b] @ W1[e] + b1[e])   (bf16 intermediate in ws)
//   stage2: out = H1 @ W2[e] + b2[e]          (f32 output)
// bf16 WMMA (v_wmma_f32_16x16x32_bf16), f32 accumulate.
// ---------------------------------------------------------------------------

#define SEQ 512
#define HID 768
#define FFN 3072

#define BM 128
#define BN 128
#define BK 64
#define PITCH 72   // LDS row pitch in bf16 elements (144B) to spread banks

typedef __attribute__((ext_vector_type(16))) __bf16 v16bf;
typedef __attribute__((ext_vector_type(8)))  float  v8f;

union BFrag {
    uint4 u[2];
    v16bf v;
};

// round-to-nearest (ties away) f32 -> bf16: one VALU add + shift
__device__ __forceinline__ unsigned short f2bf(float f) {
    union { float f; unsigned u; } x;
    x.f = f;
    return (unsigned short)((x.u + 0x8000u) >> 16);
}

// pack two f32 -> packed bf16 dword (lo, hi)
__device__ __forceinline__ unsigned f2bf_pack(float lo, float hi) {
    union { float f; unsigned u; } a, b;
    a.f = lo; b.f = hi;
    return ((a.u + 0x8000u) >> 16) | ((b.u + 0x8000u) & 0xffff0000u);
}

// Branch-free exact-GELU: Abramowitz-Stegun 7.1.26 erf (|err| < 1.5e-7),
// straight-line VALU: rcp, exp, fmas, copysign. No EXEC divergence.
__device__ __forceinline__ float gelu_exact(float x) {
    const float u = x * 0.70710678118654752f;     // x / sqrt(2)
    const float a = fabsf(u);
    const float t = __builtin_amdgcn_rcpf(fmaf(0.3275911f, a, 1.0f));
    float p = fmaf(t, 1.061405429f, -1.453152027f);
    p = fmaf(t, p, 1.421413741f);
    p = fmaf(t, p, -0.284496736f);
    p = fmaf(t, p, 0.254829592f);
    p = p * t;
    const float e     = __expf(-a * a);           // v_exp_f32 path
    const float erf_a = fmaf(-p, e, 1.0f);
    const float erf_u = copysignf(erf_a, u);
    return 0.5f * x * (1.0f + erf_u);
}

// ---------------------------------------------------------------------------
// Stage 1: C[L,F] tile = X[L,H] @ W1[H,F], +bias, gelu, -> bf16 H1
// grid: (F/BN, L/BM, B), block: 256
// ---------------------------------------------------------------------------
__global__ __launch_bounds__(256) void moe_ffn1(
    const float* __restrict__ X, const int* __restrict__ labels,
    const float* __restrict__ W1, const float* __restrict__ b1,
    unsigned short* __restrict__ H1)
{
    __shared__ unsigned short As[BM * PITCH];
    __shared__ unsigned short Bs[BN * PITCH];

    const int tid  = threadIdx.x;
    const int lane = tid & 31;
    const int wave = tid >> 5;
    const int wm   = wave >> 2;        // 0..1 : 64-row strip
    const int wn   = wave & 3;         // 0..3 : 32-col strip
    const int nb   = blockIdx.x * BN;
    const int mb   = blockIdx.y * BM;
    const int b    = blockIdx.z;
    const int e    = labels[b];

    const float* Xb = X  + ((size_t)b * SEQ + mb) * HID;
    const float* W  = W1 + (size_t)e * HID * FFN;

    v8f acc[4][2];
#pragma unroll
    for (int i = 0; i < 4; ++i)
#pragma unroll
        for (int j = 0; j < 2; ++j)
#pragma unroll
            for (int r = 0; r < 8; ++r) acc[i][j][r] = 0.0f;

    const int lr    = lane & 15;
    const int koffA = (lane >= 16) ? 8  : 0;   // A frag: K 0-7/16-23 vs 8-15/24-31
    const int koffB = (lane >= 16) ? 8  : 0;   // B frag: K 0-7/16-23 vs 8-15/24-31

#pragma unroll 1
    for (int k0 = 0; k0 < HID; k0 += BK) {
        // ---- stage A tile: X[mb..+127][k0..+63] f32 -> bf16 LDS [M][K]
        {
            const int col4  = (tid & 15) * 4;
            const int rbase = tid >> 4;
#pragma unroll
            for (int p = 0; p < 8; ++p) {
                const int row = p * 16 + rbase;
                const float4 v = *(const float4*)(Xb + (size_t)row * HID + k0 + col4);
                uint2 d;
                d.x = f2bf_pack(v.x, v.y);
                d.y = f2bf_pack(v.z, v.w);
                *(uint2*)&As[row * PITCH + col4] = d;
            }
        }
        // ---- stage B tile: W1[k0..+63][nb..+127] f32 -> bf16 LDS [N][K]
        //      (load K-row pairs, emit packed b32 stores into transposed tile)
        {
            const int n4 = (tid & 31) * 4;     // 0..124
            const int tk = tid >> 5;           // 0..7
#pragma unroll
            for (int p = 0; p < 4; ++p) {
                const int k = (p * 8 + tk) * 2;   // even, 0..62
                const float4 v0 = *(const float4*)(W + (size_t)(k0 + k)     * FFN + nb + n4);
                const float4 v1 = *(const float4*)(W + (size_t)(k0 + k + 1) * FFN + nb + n4);
                *(unsigned*)&Bs[(n4 + 0) * PITCH + k] = f2bf_pack(v0.x, v1.x);
                *(unsigned*)&Bs[(n4 + 1) * PITCH + k] = f2bf_pack(v0.y, v1.y);
                *(unsigned*)&Bs[(n4 + 2) * PITCH + k] = f2bf_pack(v0.z, v1.z);
                *(unsigned*)&Bs[(n4 + 3) * PITCH + k] = f2bf_pack(v0.w, v1.w);
            }
        }
        __syncthreads();

#pragma unroll
        for (int kk = 0; kk < BK; kk += 32) {
            BFrag bf[2];
#pragma unroll
            for (int ns = 0; ns < 2; ++ns) {
                const unsigned short* p =
                    &Bs[(wn * 32 + ns * 16 + lr) * PITCH + kk + koffB];
                bf[ns].u[0] = *(const uint4*)p;
                bf[ns].u[1] = *(const uint4*)(p + 16);
            }
#pragma unroll
            for (int ms = 0; ms < 4; ++ms) {
                const unsigned short* p =
                    &As[(wm * 64 + ms * 16 + lr) * PITCH + kk + koffA];
                BFrag af;
                af.u[0] = *(const uint4*)p;
                af.u[1] = *(const uint4*)(p + 16);
#pragma unroll
                for (int ns = 0; ns < 2; ++ns)
                    acc[ms][ns] = __builtin_amdgcn_wmma_f32_16x16x32_bf16(
                        false, af.v, false, bf[ns].v, (short)0, acc[ms][ns],
                        false, false);
            }
        }
        __syncthreads();
    }

    // ---- epilogue: bias + gelu -> bf16 H1
    const int rhi = (lane >> 4) * 8;
    unsigned short* Hb = H1 + ((size_t)b * SEQ + mb) * FFN;
#pragma unroll
    for (int ns = 0; ns < 2; ++ns) {
        const int col  = nb + wn * 32 + ns * 16 + lr;
        const float bias = b1[(size_t)e * FFN + col];
#pragma unroll
        for (int ms = 0; ms < 4; ++ms) {
#pragma unroll
            for (int r = 0; r < 8; ++r) {
                const int row = wm * 64 + ms * 16 + rhi + r;
                Hb[(size_t)row * FFN + col] = f2bf(gelu_exact(acc[ms][ns][r] + bias));
            }
        }
    }
}

// ---------------------------------------------------------------------------
// Stage 2: out[L,H] tile = H1[L,F](bf16) @ W2[F,H], +bias -> f32
// grid: (H/BN, L/BM, B), block: 256
// ---------------------------------------------------------------------------
__global__ __launch_bounds__(256) void moe_ffn2(
    const unsigned short* __restrict__ H1, const int* __restrict__ labels,
    const float* __restrict__ W2, const float* __restrict__ b2,
    float* __restrict__ out)
{
    __shared__ unsigned short As[BM * PITCH];
    __shared__ unsigned short Bs[BN * PITCH];

    const int tid  = threadIdx.x;
    const int lane = tid & 31;
    const int wave = tid >> 5;
    const int wm   = wave >> 2;
    const int wn   = wave & 3;
    const int nb   = blockIdx.x * BN;
    const int mb   = blockIdx.y * BM;
    const int b    = blockIdx.z;
    const int e    = labels[b];

    const unsigned short* Hb = H1 + ((size_t)b * SEQ + mb) * FFN;
    const float* W = W2 + (size_t)e * FFN * HID;

    v8f acc[4][2];
#pragma unroll
    for (int i = 0; i < 4; ++i)
#pragma unroll
        for (int j = 0; j < 2; ++j)
#pragma unroll
            for (int r = 0; r < 8; ++r) acc[i][j][r] = 0.0f;

    const int lr    = lane & 15;
    const int koffA = (lane >= 16) ? 8  : 0;
    const int koffB = (lane >= 16) ? 8  : 0;

#pragma unroll 1
    for (int k0 = 0; k0 < FFN; k0 += BK) {
        // ---- stage A tile: H1 (already bf16), straight copy [M][K]
        {
            const int col4  = (tid & 15) * 4;
            const int rbase = tid >> 4;
#pragma unroll
            for (int p = 0; p < 8; ++p) {
                const int row = p * 16 + rbase;
                const uint2 d = *(const uint2*)(Hb + (size_t)row * FFN + k0 + col4);
                *(uint2*)&As[row * PITCH + col4] = d;
            }
        }
        // ---- stage B tile: W2[k0..+63][nb..+127] f32 -> bf16 LDS [N][K]
        {
            const int n4 = (tid & 31) * 4;
            const int tk = tid >> 5;
#pragma unroll
            for (int p = 0; p < 4; ++p) {
                const int k = (p * 8 + tk) * 2;
                const float4 v0 = *(const float4*)(W + (size_t)(k0 + k)     * HID + nb + n4);
                const float4 v1 = *(const float4*)(W + (size_t)(k0 + k + 1) * HID + nb + n4);
                *(unsigned*)&Bs[(n4 + 0) * PITCH + k] = f2bf_pack(v0.x, v1.x);
                *(unsigned*)&Bs[(n4 + 1) * PITCH + k] = f2bf_pack(v0.y, v1.y);
                *(unsigned*)&Bs[(n4 + 2) * PITCH + k] = f2bf_pack(v0.z, v1.z);
                *(unsigned*)&Bs[(n4 + 3) * PITCH + k] = f2bf_pack(v0.w, v1.w);
            }
        }
        __syncthreads();

#pragma unroll
        for (int kk = 0; kk < BK; kk += 32) {
            BFrag bf[2];
#pragma unroll
            for (int ns = 0; ns < 2; ++ns) {
                const unsigned short* p =
                    &Bs[(wn * 32 + ns * 16 + lr) * PITCH + kk + koffB];
                bf[ns].u[0] = *(const uint4*)p;
                bf[ns].u[1] = *(const uint4*)(p + 16);
            }
#pragma unroll
            for (int ms = 0; ms < 4; ++ms) {
                const unsigned short* p =
                    &As[(wm * 64 + ms * 16 + lr) * PITCH + kk + koffA];
                BFrag af;
                af.u[0] = *(const uint4*)p;
                af.u[1] = *(const uint4*)(p + 16);
#pragma unroll
                for (int ns = 0; ns < 2; ++ns)
                    acc[ms][ns] = __builtin_amdgcn_wmma_f32_16x16x32_bf16(
                        false, af.v, false, bf[ns].v, (short)0, acc[ms][ns],
                        false, false);
            }
        }
        __syncthreads();
    }

    // ---- epilogue: bias -> f32 out
    const int rhi = (lane >> 4) * 8;
    float* Ob = out + ((size_t)b * SEQ + mb) * HID;
#pragma unroll
    for (int ns = 0; ns < 2; ++ns) {
        const int col  = nb + wn * 32 + ns * 16 + lr;
        const float bias = b2[(size_t)e * HID + col];
#pragma unroll
        for (int ms = 0; ms < 4; ++ms) {
#pragma unroll
            for (int r = 0; r < 8; ++r) {
                const int row = wm * 64 + ms * 16 + rhi + r;
                Ob[(size_t)row * HID + col] = acc[ms][ns][r] + bias;
            }
        }
    }
}

extern "C" void kernel_launch(void* const* d_in, const int* in_sizes, int n_in,
                              void* d_out, int out_size, void* d_ws, size_t ws_size,
                              hipStream_t stream) {
    (void)in_sizes; (void)n_in; (void)out_size; (void)ws_size;
    const float* X      = (const float*)d_in[0];
    const int*   labels = (const int*)  d_in[1];
    const float* W1     = (const float*)d_in[2];
    const float* b1     = (const float*)d_in[3];
    const float* W2     = (const float*)d_in[4];
    const float* b2     = (const float*)d_in[5];
    float* out = (float*)d_out;

    unsigned short* H1 = (unsigned short*)d_ws;   // 32*512*3072 bf16 = 96 MB

    dim3 block(256);
    dim3 g1(FFN / BN, SEQ / BM, 32);  // (24, 4, 32)
    dim3 g2(HID / BN, SEQ / BM, 32);  // (6, 4, 32)
    moe_ffn1<<<g1, block, 0, stream>>>(X, labels, W1, b1, H1);
    moe_ffn2<<<g2, block, 0, stream>>>(H1, labels, W2, b2, out);
}
